// BMM_S8T_S8N_S8T_73710228733999
// MI455X (gfx1250) — 8-bit matrix operand order repaired and checked by reading, compiled, not run on hardware
//
#include <hip/hip_runtime.h>
#include <stdint.h>


typedef int v8i __attribute__((ext_vector_type(8)));
typedef int v4i __attribute__((ext_vector_type(4)));

#define B_DIM 32
#define M_DIM 2048
#define N_DIM 2048
#define K_DIM 64

__device__ __forceinline__ unsigned pack4(int a, int b, int c, int d) {
    return (unsigned)((a & 0xff) | ((b & 0xff) << 8) |
                      ((c & 0xff) << 16) | ((d & 0xff) << 24));
}

__global__ __launch_bounds__(256)
void bmm_s8t_s8n_s8t_wmma(const int8_t* __restrict__ aq,
                          const int8_t* __restrict__ bq,
                          const float* __restrict__ alpha_p,
                          int8_t* __restrict__ out) {
    const int lane = threadIdx.x & 31;
    const int wave = threadIdx.x >> 5;
    const int half = lane >> 4;
    const int l16  = lane & 15;

    const int n0 = blockIdx.x * 128;
    const int m0 = blockIdx.y * 128;
    const int bz = blockIdx.z;

    const float alpha = *alpha_p;

    const int n_row = n0 + wave * 16 + l16;
    const int8_t* brow = bq + ((size_t)bz * N_DIM + n_row) * K_DIM;
    v8i afrag;
#pragma unroll
    for (int j = 0; j < 4; ++j) {
        const int* p = (const int*)(brow + half * 16 + (j & 1) * 8 + (j >> 1) * 32);
        afrag[2 * j]     = p[0];
        afrag[2 * j + 1] = p[1];
    }

    const int8_t* abase =
        aq + ((size_t)bz * M_DIM + (size_t)(m0 + l16)) * K_DIM + half * 16;
    v8i bf[8];
#pragma unroll
    for (int t = 0; t < 8; ++t) {
        const int8_t* arow = abase + (size_t)t * 16 * K_DIM;
        v4i b0 = *(const v4i*)(arow);
        v4i b1 = *(const v4i*)(arow + 32);
        bf[t][0] = b0.x; bf[t][1] = b0.y; bf[t][2] = b0.z; bf[t][3] = b0.w;
        bf[t][4] = b1.x; bf[t][5] = b1.y; bf[t][6] = b1.z; bf[t][7] = b1.w;
    }

    v8i acc[8];
#pragma unroll
    for (int t = 0; t < 8; ++t) {
        v8i z = {};
        acc[t] = __builtin_amdgcn_wmma_i32_16x16x64_iu8(
             true, afrag,  true, bf[t], z,
             false,  false);
    }

    const size_t obatch = (size_t)bz * M_DIM * N_DIM;
    const int ncol = n0 + wave * 16 + half * 8;

#pragma unroll
    for (int t = 0; t < 8; ++t) {
        int q[8];
#pragma unroll
        for (int r = 0; r < 8; ++r) {
            float f = (float)acc[t][r] * alpha;
            f = rintf(f);
            f = fminf(fmaxf(f, -128.0f), 127.0f);
            q[r] = (int)f;
        }
        unsigned lo = pack4(q[0], q[1], q[2], q[3]);
        unsigned hi = pack4(q[4], q[5], q[6], q[7]);
        unsigned long long v = ((unsigned long long)hi << 32) | lo;

        const int m = m0 + t * 16 + l16;
        unsigned long long* dst =
            (unsigned long long*)(out + obatch + (size_t)m * N_DIM + ncol);
        __builtin_nontemporal_store(v, dst);
    }
}

extern "C" void kernel_launch(void* const* d_in, const int* in_sizes, int n_in,
                              void* d_out, int out_size, void* d_ws, size_t ws_size,
                              hipStream_t stream) {
    const int8_t* a     = (const int8_t*)d_in[0];
    const int8_t* b     = (const int8_t*)d_in[1];
    const float*  alpha = (const float*)d_in[2];
    int8_t*       out   = (int8_t*)d_out;

    dim3 grid(N_DIM / 128, M_DIM / 128, B_DIM);
    dim3 block(256);
    hipLaunchKernelGGL(bmm_s8t_s8n_s8t_wmma, grid, block, 0, stream,
                       a, b, alpha, out);
}
